// hgcn_40630390620310
// MI455X (gfx1250) — matrix operand order repaired and checked by reading, not compiled, not run on hardware
//
#include <hip/hip_runtime.h>
#include <hip/hip_bf16.h>

typedef __attribute__((ext_vector_type(16))) _Float16 v16h;
typedef __attribute__((ext_vector_type(8)))  float    v8f;

constexpr int kNC = 10;   // clips (graph nodes)
constexpr int kNS = 7;    // scenes
constexpr float kInvSqrt128 = 0.08838834764831845f;

// ---------------- overlapped LDS arena (65536 B) ----------------
// @0      (32KB) sXc  f16 [16][1024] col-major   (ph0-1)
//                 UNION: sQc f16[16][128]@0, sKc @4096      (ph4-5, ph8-9a)
//                 UNION: sHsC f16 [16][512] @0              (ph6-7 hs, ph9b-10 hd)
//                 + sYs f32 [256][16] row-major @16384      (ph7-11)
// @32768  (16KB) sStateC f32 [16][256] col-major (ph1-2)  UNION sSceneC (ph11-12)
// @49152  (16KB) sU f16 [512][16] row-major      (ph1-3)  UNION sVC f32[16][128] (ph12-13)
// Non-overlapping buffers are separate __shared__ objects (exact alias info).
constexpr int OFF_QC    = 0;
constexpr int OFF_KC    = 4096;
constexpr int OFF_YS    = 16384;
constexpr int OFF_STATE = 32768;
constexpr int OFF_U     = 49152;
constexpr int OFF_VC    = 49152;
constexpr int ARENA_SIZE = 65536;

// ---------------- WMMA wrapper ----------------
static __device__ __forceinline__ v8f wmma_f16(v16h a, v16h b, v8f c) {
  return __builtin_amdgcn_wmma_f32_16x16x32_f16(false, a, false, b, (short)0, c,
                                                false, false);
}

// ---------------- fragment builders (all unconditional) ----------------
// A (16x32 f16) from global row-major f16 weights; K multiple of 32, rows padded.
static __device__ __forceinline__ v16h
load_a_g(const _Float16* __restrict__ W, int ld, int r0, int k0, int lane) {
  int r  = r0 + (lane & 15);
  int kb = k0 + ((lane >> 4) * 8);
  const _Float16* p = W + r * ld + kb;
  v16h a;
#pragma unroll
  for (int i = 0; i < 8; ++i) {
    a[i]     = p[i];        // contiguous 16B chunk  -> global_load_b128
    a[i + 8] = p[i + 16];   // contiguous 16B chunk
  }
  return a;
}

// A (16x32 f16) from LDS row-major, arbitrary ld, K multiple of 32 (e.g. Q^T).
static __device__ __forceinline__ v16h
load_a_lds_rm(const _Float16* S, int ld, int k0, int lane) {
  int r  = lane & 15;
  int kb = k0 + ((lane >> 4) * 8);
  const _Float16* p = S + r * ld + kb;
  v16h a;
#pragma unroll
  for (int i = 0; i < 8; ++i) {
    a[i]     = p[i];        // ds_load_b128
    a[i + 8] = p[i + 16];
  }
  return a;
}

// A (16x32 f16) from LDS row-major [rows x 16]: K(=position)<=16, upper half zero.
static __device__ __forceinline__ v16h
load_a_rm16(const _Float16* S, int r0, int lane) {
  int r  = r0 + (lane & 15);
  int kb = (lane >> 4) * 8;
  const _Float16* p = S + r * 16 + kb;
  v16h a;
#pragma unroll
  for (int i = 0; i < 8; ++i) {
    a[i]     = p[i];        // ds_load_b128
    a[i + 8] = (_Float16)0.0f;
  }
  return a;
}

// A (16x32 f16) from LDS col-major tile [16 cols][rowstride]: K = column index.
static __device__ __forceinline__ v16h
load_a_cm16(const _Float16* S, int rowstride, int r0, int lane) {
  int r  = r0 + (lane & 15);
  int kb = (lane >> 4) * 8;
  v16h a;
#pragma unroll
  for (int i = 0; i < 8; ++i) {
    a[i]     = S[(kb + i) * rowstride + r];
    a[i + 8] = (_Float16)0.0f;
  }
  return a;
}

// B (32x16 f16) from LDS col-major f16: lane n=lane&15, two 8-half K chunks (kb, kb+16).
static __device__ __forceinline__ v16h
load_b_cm_h(const _Float16* S, int rowstride, int cbase, int lane) {
  int n  = lane & 15;
  int kb = (lane >> 4) * 8;
  const _Float16* p = S + n * rowstride + cbase + kb;
  v16h b;
#pragma unroll
  for (int i = 0; i < 8; ++i) { b[i] = p[i]; b[i + 8] = p[i + 16]; }   // 2x ds_load_b128
  return b;
}

// B (32x16 f16) from LDS col-major f32 (zero-padded rows).
static __device__ __forceinline__ v16h
load_b_cm_f(const float* S, int rowstride, int cbase, int lane) {
  int n  = lane & 15;
  int kb = (lane >> 4) * 8;
  const float* p = S + n * rowstride + cbase + kb;
  v16h b;
#pragma unroll
  for (int i = 0; i < 8; ++i) { b[i] = (_Float16)p[i]; b[i + 8] = (_Float16)p[i + 16]; }
  return b;
}

// D store, col-major f16: 8 contiguous halfs per lane -> ds_store_b128.
// Bias preloaded (batched ds loads, one wait), applied branch-free with 0/1 mask.
static __device__ __forceinline__ void
store_cm_f16(_Float16* S, int rowstride, int r0, v8f d, const float* biasLds,
             bool dorelu, int lane) {
  int n  = lane & 15;
  int m0 = (lane >> 4) * 8;
  _Float16* p = S + n * rowstride + r0 + m0;
  float bb[8];
  if (biasLds) {
    const float* bp = biasLds + r0 + m0;
#pragma unroll
    for (int i = 0; i < 8; ++i) bb[i] = bp[i];
  } else {
#pragma unroll
    for (int i = 0; i < 8; ++i) bb[i] = 0.0f;
  }
  float bm = (n < kNC) ? 1.0f : 0.0f;
#pragma unroll
  for (int i = 0; i < 8; ++i) {
    float v = fmaf(bm, bb[i], d[i]);
    if (dorelu) v = fmaxf(v, 0.0f);
    p[i] = (_Float16)v;
  }
}

static __device__ __forceinline__ void
store_cm_f32(float* S, int rowstride, int r0, v8f d, const float* biasLds,
             bool dorelu, int lane) {
  int n  = lane & 15;
  int m0 = (lane >> 4) * 8;
  float* p = S + n * rowstride + r0 + m0;
  float bb[8];
  if (biasLds) {
    const float* bp = biasLds + r0 + m0;
#pragma unroll
    for (int i = 0; i < 8; ++i) bb[i] = bp[i];
  } else {
#pragma unroll
    for (int i = 0; i < 8; ++i) bb[i] = 0.0f;
  }
  float bm = (n < kNC) ? 1.0f : 0.0f;
#pragma unroll
  for (int i = 0; i < 8; ++i) {
    float v = fmaf(bm, bb[i], d[i]);
    if (dorelu) v = fmaxf(v, 0.0f);
    p[i] = v;
  }
}

// D store, row-major (scatter): for tiles consumed as A operands / scalar code.
static __device__ __forceinline__ void
store_rm_f16(_Float16* S, int r0, v8f d, int lane) {
  int n  = lane & 15;
  int m0 = (lane >> 4) * 8;
#pragma unroll
  for (int i = 0; i < 8; ++i) S[(r0 + m0 + i) * 16 + n] = (_Float16)d[i];
}

static __device__ __forceinline__ void
store_rm_f32(float* S, int r0, v8f d, const float* biasLds, bool dorelu, int lane) {
  int n  = lane & 15;
  int m0 = (lane >> 4) * 8;
  float bb[8];
  if (biasLds) {
    const float* bp = biasLds + r0 + m0;
#pragma unroll
    for (int i = 0; i < 8; ++i) bb[i] = bp[i];
  } else {
#pragma unroll
    for (int i = 0; i < 8; ++i) bb[i] = 0.0f;
  }
  float bm = (n < kNC) ? 1.0f : 0.0f;
#pragma unroll
  for (int i = 0; i < 8; ++i) {
    float v = fmaf(bm, bb[i], d[i]);
    if (dorelu) v = fmaxf(v, 0.0f);
    S[(r0 + m0 + i) * 16 + n] = v;
  }
}

// attention adjacency via WMMA: Bm = Q^T K (M=N=10 padded 16, K=128 -> 4 WMMAs),
// then epilogue band + alpha*Bm/sqrt(128) + C, then row softmax.
// Qc,Kc col-major [16 pos][128 chan]; A col-major [16][32], zero-padded.
static __device__ __forceinline__ void
att_softmax(const _Float16* Qc, const _Float16* Kc, float alpha,
            const float* __restrict__ Cmat, float* A, int lane) {
  v8f acc = {};
#pragma unroll
  for (int ks = 0; ks < 4; ++ks) {
    v16h af = load_a_lds_rm(Qc, 128, ks * 32, lane);   // A = Q^T (rows contiguous)
    v16h bf = load_b_cm_h(Kc, 128, ks * 32, lane);     // B = K
    acc = wmma_f16(af, bf, acc);
  }
  // D[n=row, m=col]: lane holds col m, rows n = m0..m0+7
  int m  = lane & 15;
  int m0 = (lane >> 4) * 8;
  int mc = (m < kNC) ? m : (kNC - 1);                  // clamp -> uncond loads
#pragma unroll
  for (int i = 0; i < 8; ++i) {
    int n  = m0 + i;
    int nc = (n < kNC) ? n : (kNC - 1);
    float cv = Cmat[nc * kNC + mc];
    int dd = n - m; dd = dd < 0 ? -dd : dd;
    float band = (dd <= 3) ? 1.0f : 0.0f;              // A_shot = band_adj(10, 3)
    float v = band + alpha * acc[i] * kInvSqrt128 + cv;
    if ((n < kNC) && (m < kNC)) A[m * 32 + n] = v;     // contiguous over i
  }
  __syncthreads();
  if (lane < kNC) {                                    // softmax over m (row n=lane)
    float mx = -1e30f;
#pragma unroll
    for (int mm = 0; mm < kNC; ++mm) mx = fmaxf(mx, A[mm * 32 + lane]);
    float e[kNC];
    float sum = 0.0f;
#pragma unroll
    for (int mm = 0; mm < kNC; ++mm) {
      e[mm] = __expf(A[mm * 32 + lane] - mx);
      sum += e[mm];
    }
    float inv = 1.0f / sum;
#pragma unroll
    for (int mm = 0; mm < kNC; ++mm) A[mm * 32 + lane] = e[mm] * inv;
  }
  __syncthreads();
}

// ---------------- weight f32 -> f16 conversion ----------------
__global__ void cvt_f32_f16_kernel(const float* __restrict__ src,
                                   _Float16* __restrict__ dst, int n) {
  int i = blockIdx.x * blockDim.x + threadIdx.x;
  if (i < n) dst[i] = (_Float16)src[i];
}

__global__ void cvt_pad_f16_kernel(const float* __restrict__ src,
                                   _Float16* __restrict__ dst, int nsrc, int ntot) {
  int i = blockIdx.x * blockDim.x + threadIdx.x;
  if (i < ntot) dst[i] = (i < nsrc) ? (_Float16)src[i] : (_Float16)0.0f;
}

// ---------------- fused per-batch pipeline: one wave per batch ----------------
__global__ __launch_bounds__(32)
void hgcn_main(const float* __restrict__ x, const float* __restrict__ esp,
               const _Float16* __restrict__ w_state, const float* __restrict__ b_state,
               const _Float16* __restrict__ w_info,  const float* __restrict__ b_info,
               const _Float16* __restrict__ wq_s, const _Float16* __restrict__ wk_s,
               const float* __restrict__ C_s, const float* __restrict__ alpha_s,
               const _Float16* __restrict__ w_sagg, const float* __restrict__ b_sagg,
               const _Float16* __restrict__ wq_d, const _Float16* __restrict__ wk_d,
               const float* __restrict__ C_d, const float* __restrict__ alpha_d,
               const _Float16* __restrict__ w_dagg, const float* __restrict__ b_dagg,
               const float* __restrict__ S_d, const float* __restrict__ alpha_diff,
               const _Float16* __restrict__ w_mot, const float* __restrict__ b_mot,
               const float* __restrict__ w_mean, const float* __restrict__ b_mean,
               const float* __restrict__ w_logvar, const float* __restrict__ b_logvar,
               float* __restrict__ out0, float* __restrict__ out_mu,
               float* __restrict__ out_std) {
  __shared__ __align__(16) unsigned char arena[ARENA_SIZE];
  __shared__ __align__(16) _Float16 sShotC[16 * 512];  // col-major (ph3-10)
  __shared__ __align__(16) float sAsta[16 * 32];       // col-major, zero-padded
  __shared__ __align__(16) float sAatt[16 * 32];       // col-major, zero-padded
  __shared__ __align__(16) float sT[16 * 16];          // row-major, zero-padded
  __shared__ __align__(16) float sBias[1152];          // b_state|b_info|b_sagg|b_mot
  __shared__ float sSmall[8];

  _Float16* sXc    = (_Float16*)(arena);
  _Float16* sQc    = (_Float16*)(arena + OFF_QC);
  _Float16* sKc    = (_Float16*)(arena + OFF_KC);
  _Float16* sHsC   = (_Float16*)(arena);
  float*    sYs    = (float*)(arena + OFF_YS);
  float*    sStateC= (float*)(arena + OFF_STATE);
  float*    sSceneC= (float*)(arena + OFF_STATE);
  _Float16* sU     = (_Float16*)(arena + OFF_U);
  float*    sVC    = (float*)(arena + OFF_VC);

  const int lane = threadIdx.x;            // wave32, blockDim.x == 32
  const int b    = blockIdx.x;
  const float* xb = x + (size_t)b * (1024 * kNC);

  __builtin_prefetch(xb + lane * 320, 0, 0);     // global_prefetch toward L2

  // ---- Phase 0a: zero padded regions + stage biases into LDS
  {
    float4 zz; zz.x = zz.y = zz.z = zz.w = 0.0f;
    float4* z0 = (float4*)arena;                        // sXc region, 32768 B
    for (int i = lane; i < 2048; i += 32) z0[i] = zz;
    float4* z1 = (float4*)sAsta;
    for (int i = lane; i < 128; i += 32) z1[i] = zz;
    float4* z2 = (float4*)sAatt;
    for (int i = lane; i < 128; i += 32) z2[i] = zz;
    float4* z3 = (float4*)sT;
    for (int i = lane; i < 64; i += 32) z3[i] = zz;
  }
  for (int i = lane; i < 256; i += 32) sBias[i]        = b_state[i];
  for (int i = lane; i < 512; i += 32) sBias[256 + i]  = b_info[i];
  for (int i = lane; i < 256; i += 32) sBias[768 + i]  = b_sagg[i];
  for (int i = lane; i < 128; i += 32) sBias[1024 + i] = b_mot[i];
  __syncthreads();

  // ---- Phase 0b: stage x[b] -> LDS f16 col-major [16 pos][1024 chan]
  for (int c0 = 0; c0 < 1024; c0 += 32) {
    int c = c0 + lane;
    const float2* p2 = (const float2*)(xb + (size_t)c * kNC);  // 8B aligned
    float2 v0 = p2[0], v1 = p2[1], v2 = p2[2], v3 = p2[3], v4 = p2[4];
    sXc[0 * 1024 + c] = (_Float16)v0.x;  sXc[1 * 1024 + c] = (_Float16)v0.y;
    sXc[2 * 1024 + c] = (_Float16)v1.x;  sXc[3 * 1024 + c] = (_Float16)v1.y;
    sXc[4 * 1024 + c] = (_Float16)v2.x;  sXc[5 * 1024 + c] = (_Float16)v2.y;
    sXc[6 * 1024 + c] = (_Float16)v3.x;  sXc[7 * 1024 + c] = (_Float16)v3.y;
    sXc[8 * 1024 + c] = (_Float16)v4.x;  sXc[9 * 1024 + c] = (_Float16)v4.y;
  }
  __syncthreads();

  // ---- Phase 1: state_pre = W_state x, U = W_info x (grouped g=8)
  for (int g = 0; g < 8; ++g) {
    v8f accS[2] = {v8f{}, v8f{}};
    v8f accU[4] = {v8f{}, v8f{}, v8f{}, v8f{}};
    for (int ks = 0; ks < 4; ++ks) {
      v16h bf = load_b_cm_h(sXc, 1024, g * 128 + ks * 32, lane);
#pragma unroll
      for (int mt = 0; mt < 2; ++mt)
        accS[mt] = wmma_f16(load_a_g(w_state, 128, g * 32 + mt * 16, ks * 32, lane),
                            bf, accS[mt]);
#pragma unroll
      for (int mt = 0; mt < 4; ++mt)
        accU[mt] = wmma_f16(load_a_g(w_info, 128, g * 64 + mt * 16, ks * 32, lane),
                            bf, accU[mt]);
    }
#pragma unroll
    for (int mt = 0; mt < 2; ++mt)
      store_cm_f32(sStateC, 256, g * 32 + mt * 16, accS[mt], sBias + 0, true, lane);
#pragma unroll
    for (int mt = 0; mt < 4; ++mt)
      store_rm_f16(sU, g * 64 + mt * 16, accU[mt], lane);   // padding cols = 0
  }
  __syncthreads();

  // ---- Phase 2: A_sta[n,m] = (mean_c tanh(state[c,n]-state[c,m]) + I) * band2
  for (int p = lane; p < kNC * kNC; p += 32) {
    int n = p / kNC, m = p % kNC;
    float s = 0.0f;
    for (int c = 0; c < 256; ++c) {
      float d = sStateC[n * 256 + c] - sStateC[m * 256 + c];
      d = fminf(fmaxf(d, -15.0f), 15.0f);
      float e = __expf(2.0f * d);
      s += (e - 1.0f) / (e + 1.0f);
    }
    float a = s * (1.0f / 256.0f) + ((n == m) ? 1.0f : 0.0f);
    int dnm = n - m; dnm = dnm < 0 ? -dnm : dnm;
    sAsta[m * 32 + n] = (dnm <= 2) ? a : 0.0f;      // A_clip = band_adj(10, 2)
  }
  __syncthreads();

  // ---- Phase 3: shot = relu(U @ A_sta + b_info) -> col-major [16][512]
  {
    v16h bA = load_b_cm_f(sAsta, 32, 0, lane);
    for (int mt = 0; mt < 32; ++mt) {
      v8f acc = {};
      acc = wmma_f16(load_a_rm16(sU, mt * 16, lane), bA, acc);
      store_cm_f16(sShotC, 512, mt * 16, acc, sBias + 256, true, lane);
    }
  }
  __syncthreads();

  // ---- Phase 4: shotGCN q/k projections (grouped g=8, 512->128)
  for (int g = 0; g < 8; ++g) {
    v8f aq = {}, ak = {};
    for (int ks = 0; ks < 2; ++ks) {
      v16h bf = load_b_cm_h(sShotC, 512, g * 64 + ks * 32, lane);
      aq = wmma_f16(load_a_g(wq_s, 64, g * 16, ks * 32, lane), bf, aq);
      ak = wmma_f16(load_a_g(wk_s, 64, g * 16, ks * 32, lane), bf, ak);
    }
    store_cm_f16(sQc, 128, g * 16, aq, nullptr, false, lane);
    store_cm_f16(sKc, 128, g * 16, ak, nullptr, false, lane);
  }
  __syncthreads();

  // ---- Phase 5: As = softmax(A_shot + alpha_s*QK/sqrt(128) + C_s)  [WMMA]
  att_softmax(sQc, sKc, alpha_s[0], C_s, sAatt, lane);

  // ---- Phase 6: hs = shot @ As -> col-major [16][512]
  {
    v16h bA = load_b_cm_f(sAatt, 32, 0, lane);
    for (int mt = 0; mt < 32; ++mt) {
      v8f acc = {};
      acc = wmma_f16(load_a_cm16(sShotC, 512, mt * 16, lane), bA, acc);
      store_cm_f16(sHsC, 512, mt * 16, acc, nullptr, false, lane);
    }
  }
  __syncthreads();

  // ---- Phase 7: ys = relu(gconv(hs, w_sagg, g=8) + b_sagg)  [256 x 10]
  for (int g = 0; g < 8; ++g) {
    v8f acc[2] = {v8f{}, v8f{}};
    for (int ks = 0; ks < 2; ++ks) {
      v16h bf = load_b_cm_h(sHsC, 512, g * 64 + ks * 32, lane);
#pragma unroll
      for (int mt = 0; mt < 2; ++mt)
        acc[mt] = wmma_f16(load_a_g(w_sagg, 64, g * 32 + mt * 16, ks * 32, lane),
                           bf, acc[mt]);
    }
#pragma unroll
    for (int mt = 0; mt < 2; ++mt)
      store_rm_f32(sYs, g * 32 + mt * 16, acc[mt], sBias + 768, true, lane);
  }
  __syncthreads();

  // ---- Phase 8: diffGCN dense q/k (512->128): 1 B-fragment feeds 16 WMMAs
  {
    v8f aq[8], ak[8];
#pragma unroll
    for (int mt = 0; mt < 8; ++mt) { aq[mt] = v8f{}; ak[mt] = v8f{}; }
    for (int ks = 0; ks < 16; ++ks) {
      v16h bf = load_b_cm_h(sShotC, 512, ks * 32, lane);
#pragma unroll
      for (int mt = 0; mt < 8; ++mt) {
        aq[mt] = wmma_f16(load_a_g(wq_d, 512, mt * 16, ks * 32, lane), bf, aq[mt]);
        ak[mt] = wmma_f16(load_a_g(wk_d, 512, mt * 16, ks * 32, lane), bf, ak[mt]);
      }
    }
#pragma unroll
    for (int mt = 0; mt < 8; ++mt) {
      store_cm_f16(sQc, 128, mt * 16, aq[mt], nullptr, false, lane);
      store_cm_f16(sKc, 128, mt * 16, ak[mt], nullptr, false, lane);
    }
  }
  __syncthreads();

  // ---- Phase 9: Ad = softmax(...) [WMMA]; hd = shot @ Ad -> sHsC
  att_softmax(sQc, sKc, alpha_d[0], C_d, sAatt, lane);
  {
    v16h bA = load_b_cm_f(sAatt, 32, 0, lane);
    for (int mt = 0; mt < 32; ++mt) {
      v8f acc = {};
      acc = wmma_f16(load_a_cm16(sShotC, 512, mt * 16, lane), bA, acc);
      store_cm_f16(sHsC, 512, mt * 16, acc, nullptr, false, lane);   // hd
    }
  }
  __syncthreads();

  // ---- Phase 10: T = S_d + alpha_diff * relu(w_dagg @ hd + b_dagg)  [7 x 10]
  {
    v8f acc = {};
    for (int ks = 0; ks < 16; ++ks) {
      v16h bf = load_b_cm_h(sHsC, 512, ks * 32, lane);
      acc = wmma_f16(load_a_g(w_dagg, 512, 0, ks * 32, lane), bf, acc);
    }
    float adiff = alpha_diff[0];
    int n  = lane & 15;
    int m0 = (lane >> 4) * 8;
    int nc = (n < kNC) ? n : (kNC - 1);      // clamp -> unconditional loads
#pragma unroll
    for (int i = 0; i < 8; ++i) {
      int r  = m0 + i;
      int rc = (r < kNS) ? r : (kNS - 1);
      float sv = S_d[rc * kNC + nc];
      float bv = b_dagg[rc];
      float tv = sv + adiff * fmaxf(acc[i] + bv, 0.0f);
      if ((n < kNC) && (r < kNS)) sT[r * 16 + n] = tv;   // guarded store only
    }
  }
  __syncthreads();

  // ---- Phase 11: scene[c,m] = sum_n ys[c,n] * T[m,n] -> col-major [16][256]
  for (int p = lane; p < 16 * 256; p += 32) {
    int m = p >> 8, c = p & 255;
    float s = 0.0f;
#pragma unroll
    for (int n = 0; n < kNC; ++n) s += sYs[c * 16 + n] * sT[m * 16 + n];
    sSceneC[m * 256 + c] = s;          // cols >= 7 exactly zero (T rows >=7 zero)
  }
  __syncthreads();

  // ---- Phase 12: v = relu(gconv(scene, w_mot, g=8) + b_mot)  [128 x 7]
  for (int g = 0; g < 8; ++g) {
    v16h bf = load_b_cm_f(sSceneC, 256, g * 32, lane);
    v8f acc = {};
    acc = wmma_f16(load_a_g(w_mot, 32, g * 16, 0, lane), bf, acc);
    store_cm_f32(sVC, 128, g * 16, acc, sBias + 1024, true, lane);
  }
  __syncthreads();

  // ---- Phase 13: heads + reparameterize
  if (lane < kNS) {
    float mu = b_mean[0], lv = b_logvar[0];
    for (int c = 0; c < 128; ++c) {
      float vv = sVC[lane * 128 + c];
      mu += w_mean[c] * vv;
      lv += w_logvar[c] * vv;
    }
    float sd = __expf(0.5f * lv);
    out_mu[b * kNS + lane]  = mu;
    out_std[b * kNS + lane] = sd;
    sSmall[lane] = mu + sd * esp[b * kNS + lane];
  }
  __syncthreads();
  if (lane == 0) {
    float s = 0.0f;
#pragma unroll
    for (int m = 0; m < kNS; ++m) s += sSmall[m];
    out0[b] = s;                       // mean(scores) * NS == sum(scores)
  }
}

// ---------------- host launcher ----------------
extern "C" void kernel_launch(void* const* d_in, const int* in_sizes, int n_in,
                              void* d_out, int out_size, void* d_ws, size_t ws_size,
                              hipStream_t stream) {
  (void)n_in; (void)out_size; (void)ws_size;

  const float* x        = (const float*)d_in[0];
  const float* esp      = (const float*)d_in[1];
  const float* w_state  = (const float*)d_in[2];
  const float* b_state  = (const float*)d_in[3];
  const float* w_info   = (const float*)d_in[4];
  const float* b_info   = (const float*)d_in[5];
  const float* wq_s     = (const float*)d_in[6];
  const float* wk_s     = (const float*)d_in[7];
  const float* C_s      = (const float*)d_in[8];
  const float* alpha_s  = (const float*)d_in[9];
  const float* w_sagg   = (const float*)d_in[10];
  const float* b_sagg   = (const float*)d_in[11];
  const float* wq_d     = (const float*)d_in[12];
  const float* wk_d     = (const float*)d_in[13];
  const float* C_d      = (const float*)d_in[14];
  const float* alpha_d  = (const float*)d_in[15];
  const float* w_dagg   = (const float*)d_in[16];
  const float* b_dagg   = (const float*)d_in[17];
  const float* S_d      = (const float*)d_in[18];
  const float* alpha_df = (const float*)d_in[19];
  const float* w_mot    = (const float*)d_in[20];
  const float* b_mot    = (const float*)d_in[21];
  const float* w_mean   = (const float*)d_in[22];
  const float* b_mean   = (const float*)d_in[23];
  const float* w_logvar = (const float*)d_in[24];
  const float* b_logvar = (const float*)d_in[25];

  const int B = in_sizes[0] / (1024 * 10);

  // carve f16 weight copies out of workspace
  _Float16* ws = (_Float16*)d_ws;
  size_t off = 0;
  auto carve = [&](int n) { _Float16* p = ws + off; off += (size_t)n; return p; };
  _Float16* hw_state = carve(256 * 128);
  _Float16* hw_info  = carve(512 * 128);
  _Float16* hwq_s    = carve(128 * 64);
  _Float16* hwk_s    = carve(128 * 64);
  _Float16* hw_sagg  = carve(256 * 64);
  _Float16* hwq_d    = carve(128 * 512);
  _Float16* hwk_d    = carve(128 * 512);
  _Float16* hw_dagg  = carve(16 * 512);      // padded 7 -> 16 rows
  _Float16* hw_mot   = carve(128 * 32);

  auto cvt = [&](const float* s, _Float16* d, int n) {
    cvt_f32_f16_kernel<<<(n + 255) / 256, 256, 0, stream>>>(s, d, n);
  };
  cvt(w_state, hw_state, 256 * 128);
  cvt(w_info,  hw_info,  512 * 128);
  cvt(wq_s,    hwq_s,    128 * 64);
  cvt(wk_s,    hwk_s,    128 * 64);
  cvt(w_sagg,  hw_sagg,  256 * 64);
  cvt(wq_d,    hwq_d,    128 * 512);
  cvt(wk_d,    hwk_d,    128 * 512);
  cvt(w_mot,   hw_mot,   128 * 32);
  cvt_pad_f16_kernel<<<(16 * 512 + 255) / 256, 256, 0, stream>>>(
      w_dagg, hw_dagg, 7 * 512, 16 * 512);

  float* out0    = (float*)d_out;           // [B]
  float* out_mu  = out0 + B;                // [B,7]
  float* out_std = out0 + B + B * 7;        // [B,7]

  hgcn_main<<<dim3(B), dim3(32), 0, stream>>>(
      x, esp,
      hw_state, b_state, hw_info, b_info,
      hwq_s, hwk_s, C_s, alpha_s, hw_sagg, b_sagg,
      hwq_d, hwk_d, C_d, alpha_d, hw_dagg, b_dagg, S_d, alpha_df,
      hw_mot, b_mot, w_mean, b_mean, w_logvar, b_logvar,
      out0, out_mu, out_std);
}
